// DualBranchModel_88467736363132
// MI455X (gfx1250) — compile-verified, matrix operand order checked by reading, not run on hardware
//
#include <hip/hip_runtime.h>
#include <hip/hip_bf16.h>
#include <math.h>

// ---- model dims ----
#define Ncfg  400
#define Dcfg  256
#define NHcfg 4
#define NCcfg 2
#define Bcfg  64
#define NLcfg 2
#define HDcfg 64
#define D3cfg (3*Dcfg)
#define D4cfg (4*Dcfg)

typedef __bf16 v16bf __attribute__((ext_vector_type(16)));
typedef float  v8f   __attribute__((ext_vector_type(8)));

// Pack two contiguous 8-float chunks (p[8h..8h+7], p[16+8h..16+8h+7]) into a
// 16-element bf16 fragment, matching CDNA5 16-bit 16x32 A/B VGPR layout.
__device__ __forceinline__ v16bf load_frag16(const float* __restrict__ p, int half)
{
    const float4* q0 = (const float4*)(p + 8 * half);
    const float4* q1 = (const float4*)(p + 16 + 8 * half);
    float4 u0 = q0[0], u1 = q0[1];
    float4 u2 = q1[0], u3 = q1[1];
    v16bf f;
    f[0]  = (__bf16)u0.x; f[1]  = (__bf16)u0.y; f[2]  = (__bf16)u0.z; f[3]  = (__bf16)u0.w;
    f[4]  = (__bf16)u1.x; f[5]  = (__bf16)u1.y; f[6]  = (__bf16)u1.z; f[7]  = (__bf16)u1.w;
    f[8]  = (__bf16)u2.x; f[9]  = (__bf16)u2.y; f[10] = (__bf16)u2.z; f[11] = (__bf16)u2.w;
    f[12] = (__bf16)u3.x; f[13] = (__bf16)u3.y; f[14] = (__bf16)u3.z; f[15] = (__bf16)u3.w;
    return f;
}

// fragment element e corresponds to k-offset within the 32-step:
//   e<8 : 8*half + e     ; e>=8 : 8 + 8*half + e
__device__ __forceinline__ int kmap(int e, int half)
{
    return (e < 8) ? (8 * half + e) : (8 + 8 * half + e);
}

// ---------------------------------------------------------------------------
// Generic batched GEMM, bf16 WMMA, f32 accumulate, NT=4 tiles (16x64) / wave.
//   C[z][m,n] = act( scale * sum_k A[z][m,k] * B[z][k,n] + bias[n] )
// z = blockIdx.z -> (b = z/H, h = z%H); offsets b*s?b + h*s?h.
// A row-major (lda). B element at k*ldbk + n*ldbn.
// ---------------------------------------------------------------------------
__global__ __launch_bounds__(32)
void gemm_wmma_bf16(const float* __restrict__ A, long long sAb, long long sAh, int lda,
                    const float* __restrict__ Bp, long long sBb, long long sBh,
                    long long ldbk, long long ldbn,
                    const float* __restrict__ bias,
                    float* __restrict__ C, long long sCb, long long sCh, int ldc,
                    int M, int Nout, int K, int H, float scale, int relu)
{
    constexpr int NT = 4;
    int z = blockIdx.z;
    int b = z / H, hh = z - b * H;
    A  += (size_t)b * sAb + (size_t)hh * sAh;
    Bp += (size_t)b * sBb + (size_t)hh * sBh;
    C  += (size_t)b * sCb + (size_t)hh * sCh;

    int lane = threadIdx.x;            // 0..31
    int half = lane >> 4;              // 0 or 1
    int idx  = lane & 15;

    int m  = blockIdx.x * 16 + idx;
    int mr = (m < M) ? m : (M - 1);    // clamped read row; store guarded
    const float* __restrict__ arow = A + (size_t)mr * lda;

    int n[NT], nr[NT];
#pragma unroll
    for (int t = 0; t < NT; ++t) {
        n[t]  = blockIdx.y * (16 * NT) + 16 * t + idx;
        nr[t] = (n[t] < Nout) ? n[t] : (Nout - 1);
    }

    v8f zero = {};
    v8f acc[NT];
#pragma unroll
    for (int t = 0; t < NT; ++t) acc[t] = zero;

    const int K32 = K & ~31;

    if (ldbk == 1) {
        // B contiguous along k: fully vectorized b128 path
        for (int k0 = 0; k0 < K32; k0 += 32) {
            v16bf afr = load_frag16(arow + k0, half);
#pragma unroll
            for (int t = 0; t < NT; ++t) {
                v16bf bfr = load_frag16(Bp + (size_t)nr[t] * ldbn + k0, half);
                acc[t] = __builtin_amdgcn_wmma_f32_16x16x32_bf16(
                    false, afr, false, bfr, (short)0, acc[t], false, false);
            }
        }
    } else {
        // B strided along k: unconditional scalar gathers (no exec branches)
        for (int k0 = 0; k0 < K32; k0 += 32) {
            v16bf afr = load_frag16(arow + k0, half);
#pragma unroll
            for (int t = 0; t < NT; ++t) {
                const float* bp = Bp + (size_t)nr[t] * ldbn;
                v16bf bfr;
#pragma unroll
                for (int e = 0; e < 16; ++e) {
                    int k = k0 + kmap(e, half);
                    bfr[e] = (__bf16)bp[(size_t)k * ldbk];
                }
                acc[t] = __builtin_amdgcn_wmma_f32_16x16x32_bf16(
                    false, afr, false, bfr, (short)0, acc[t], false, false);
            }
        }
    }

    // K tail (at most one iteration; K=400 cases). Clamped loads + value
    // select keeps this branch-free.
    for (int k0 = K32; k0 < K; k0 += 32) {
        v16bf afr;
#pragma unroll
        for (int e = 0; e < 16; ++e) {
            int k  = k0 + kmap(e, half);
            int kc = (k < K) ? k : (K - 1);
            float av = arow[kc];
            afr[e] = (__bf16)((k < K) ? av : 0.0f);
        }
#pragma unroll
        for (int t = 0; t < NT; ++t) {
            const float* bp = Bp + (size_t)nr[t] * ldbn;
            v16bf bfr;
#pragma unroll
            for (int e = 0; e < 16; ++e) {
                int k  = k0 + kmap(e, half);
                int kc = (k < K) ? k : (K - 1);
                float bv = bp[(size_t)kc * ldbk];
                bfr[e] = (__bf16)((k < K) ? bv : 0.0f);
            }
            acc[t] = __builtin_amdgcn_wmma_f32_16x16x32_bf16(
                false, afr, false, bfr, (short)0, acc[t], false, false);
        }
    }

    // epilogue: C/D layout = vgpr r, lanes 0-15: M=r ; lanes 16-31: M=r+8
#pragma unroll
    for (int t = 0; t < NT; ++t) {
        if (n[t] < Nout) {
            float bb = bias ? bias[n[t]] : 0.0f;
#pragma unroll
            for (int r = 0; r < 8; ++r) {
                int mrow = blockIdx.x * 16 + r + 8 * half;
                if (mrow < M) {
                    float v = acc[t][r] * scale + bb;
                    if (relu) v = fmaxf(v, 0.0f);
                    C[(size_t)mrow * ldc + n[t]] = v;
                }
            }
        }
    }
}

// ---- softmax over rows of length L (one wave per row, in place) ----
__global__ __launch_bounds__(32)
void softmax_rows(float* __restrict__ x, int R, int L)
{
    int row = blockIdx.x;
    if (row >= R) return;
    float* p = x + (size_t)row * L;
    int lane = threadIdx.x;

    float mx = -3.0e38f;
    for (int i = lane; i < L; i += 32) mx = fmaxf(mx, p[i]);
#pragma unroll
    for (int off = 16; off; off >>= 1) mx = fmaxf(mx, __shfl_xor(mx, off, 32));

    float s = 0.0f;
    for (int i = lane; i < L; i += 32) {
        float e = __expf(p[i] - mx);
        p[i] = e;
        s += e;
    }
#pragma unroll
    for (int off = 16; off; off >>= 1) s += __shfl_xor(s, off, 32);
    float inv = 1.0f / s;
    for (int i = lane; i < L; i += 32) p[i] *= inv;
}

// ---- out = LN(x + res) * g + b over rows of length D=256 (one wave/row) ----
__global__ __launch_bounds__(32)
void ln_add_rows(const float* __restrict__ x, const float* __restrict__ res,
                 const float* __restrict__ g, const float* __restrict__ bta,
                 float* __restrict__ out, int R)
{
    int row = blockIdx.x;
    if (row >= R) return;
    const float* xp = x + (size_t)row * Dcfg;
    const float* rp = res + (size_t)row * Dcfg;
    float* op = out + (size_t)row * Dcfg;
    int lane = threadIdx.x;

    float v[8];
    float s = 0.0f;
#pragma unroll
    for (int j = 0; j < 8; ++j) {
        int i = lane + 32 * j;
        v[j] = xp[i] + rp[i];
        s += v[j];
    }
#pragma unroll
    for (int off = 16; off; off >>= 1) s += __shfl_xor(s, off, 32);
    float mu = s * (1.0f / Dcfg);

    float s2 = 0.0f;
#pragma unroll
    for (int j = 0; j < 8; ++j) { float d = v[j] - mu; s2 += d * d; }
#pragma unroll
    for (int off = 16; off; off >>= 1) s2 += __shfl_xor(s2, off, 32);
    float rstd = rsqrtf(s2 * (1.0f / Dcfg) + 1e-5f);

#pragma unroll
    for (int j = 0; j < 8; ++j) {
        int i = lane + 32 * j;
        op[i] = (v[j] - mu) * rstd * g[i] + bta[i];
    }
}

// ---- dinv[b,j] = rsqrt(sum_i sc[b,i,j]) ----
__global__ void deg_dinv(const float* __restrict__ sc, float* __restrict__ dinv, int total)
{
    int t = blockIdx.x * blockDim.x + threadIdx.x;
    if (t >= total) return;
    int b = t / Ncfg, j = t - b * Ncfg;
    const float* p = sc + (size_t)b * Ncfg * Ncfg + j;
    float s = 0.0f;
    for (int i = 0; i < Ncfg; ++i) s += p[(size_t)i * Ncfg];
    dinv[t] = (s > 0.0f) ? rsqrtf(s) : 0.0f;
}

// ---- AnT[b,j,i] = dinv[b,i] * sc[b,i,j] * dinv[b,j] ----
__global__ void build_anT(const float* __restrict__ sc, const float* __restrict__ dinv,
                          float* __restrict__ anT, long long total)
{
    long long t = (long long)blockIdx.x * blockDim.x + threadIdx.x;
    if (t >= total) return;
    int i = (int)(t % Ncfg);
    long long r = t / Ncfg;
    int j = (int)(r % Ncfg);
    int b = (int)(r / Ncfg);
    float di = dinv[b * Ncfg + i];
    float dj = dinv[b * Ncfg + j];
    anT[t] = di * sc[((size_t)b * Ncfg + i) * Ncfg + j] * dj;
}

// ---- o = 0.5*(a + b) ----
__global__ void avg2(const float* __restrict__ a, const float* __restrict__ b,
                     float* __restrict__ o, long long n)
{
    long long t = (long long)blockIdx.x * blockDim.x + threadIdx.x;
    if (t < n) o[t] = 0.5f * (a[t] + b[t]);
}

// ---------------------------------------------------------------------------
// host-side helpers
// ---------------------------------------------------------------------------
static inline void gemm(hipStream_t s,
                        const float* A, long long sAb, long long sAh, int lda,
                        const float* Bp, long long sBb, long long sBh,
                        long long ldbk, long long ldbn,
                        const float* bias,
                        float* C, long long sCb, long long sCh, int ldc,
                        int M, int Nout, int K, int batch, int H,
                        float scale, int relu)
{
    dim3 grid((M + 15) / 16, (Nout + 63) / 64, batch * H);
    gemm_wmma_bf16<<<grid, 32, 0, s>>>(A, sAb, sAh, lda, Bp, sBb, sBh, ldbk, ldbn,
                                       bias, C, sCb, sCh, ldc, M, Nout, K, H, scale, relu);
}

extern "C" void kernel_launch(void* const* d_in, const int* in_sizes, int n_in,
                              void* d_out, int out_size, void* d_ws, size_t ws_size,
                              hipStream_t stream)
{
    (void)in_sizes; (void)n_in; (void)out_size; (void)ws_size;

    const float* fc        = (const float*)d_in[0];
    const float* sc        = (const float*)d_in[1];
    const float* proj_w    = (const float*)d_in[2];
    const float* proj_b    = (const float*)d_in[3];
    const float* eqkv_w    = (const float*)d_in[4];
    const float* eqkv_b    = (const float*)d_in[5];
    const float* eout_w    = (const float*)d_in[6];
    const float* eout_b    = (const float*)d_in[7];
    const float* el1_w     = (const float*)d_in[8];
    const float* el1_b     = (const float*)d_in[9];
    const float* el2_w     = (const float*)d_in[10];
    const float* el2_b     = (const float*)d_in[11];
    const float* en1_g     = (const float*)d_in[12];
    const float* en1_b     = (const float*)d_in[13];
    const float* en2_g     = (const float*)d_in[14];
    const float* en2_b     = (const float*)d_in[15];
    const float* gcn_w1    = (const float*)d_in[16];
    const float* gcn_b1    = (const float*)d_in[17];
    const float* gcn_w2    = (const float*)d_in[18];
    const float* gcn_b2    = (const float*)d_in[19];
    const float* ca1_qkv_w = (const float*)d_in[20];
    const float* ca1_qkv_b = (const float*)d_in[21];
    const float* ca1_out_w = (const float*)d_in[22];
    const float* ca1_out_b = (const float*)d_in[23];
    const float* ca1_n_g   = (const float*)d_in[24];
    const float* ca1_n_b   = (const float*)d_in[25];
    const float* ca2_qkv_w = (const float*)d_in[26];
    const float* ca2_qkv_b = (const float*)d_in[27];
    const float* ca2_out_w = (const float*)d_in[28];
    const float* ca2_out_b = (const float*)d_in[29];
    const float* ca2_n_g   = (const float*)d_in[30];
    const float* ca2_n_b   = (const float*)d_in[31];
    const float* cls_w1    = (const float*)d_in[32];
    const float* cls_b1    = (const float*)d_in[33];
    const float* cls_w2    = (const float*)d_in[34];
    const float* cls_b2    = (const float*)d_in[35];

    float* ws = (float*)d_ws;
    float* logits = (float*)d_out;

    const size_t BND = (size_t)Bcfg * Ncfg * Dcfg;          // 6,553,600
    const size_t BIG = (size_t)Bcfg * NHcfg * Ncfg * Ncfg;  // 40,960,000
    float* o_h   = ws;                      // [B,N,D]   Z_F (evolving)
    float* o_qkv = o_h   + BND;             // [B,N,3D]
    float* o_ctx = o_qkv + 3 * BND;         // [B,N,D]
    float* o_t1  = o_ctx + BND;             // [B,N,4D]  ffn mid / Z_joint
    float* o_t2  = o_t1  + 4 * BND;         // [B,N,D]
    float* o_ZS  = o_t2  + BND;             // [B,N,D]
    float* o_X   = o_ZS  + BND;             // [B,N,D]
    float* o_h1  = o_X   + BND;             // [B,N,D]
    float* o_ZFc = o_h1  + BND;             // [B,N,D]
    float* o_ZSc = o_ZFc + BND;             // [B,N,D]
    float* o_big = o_ZSc + BND;             // [B,NH,N,N] scores; reused for AnT [B,N,N]
    float* o_dnv = o_big + BIG;             // [B,N]
    float* o_cls = o_dnv + (size_t)Bcfg * Ncfg;  // [B,128]

    const long long NN  = (long long)Ncfg * Ncfg;
    const long long ND  = (long long)Ncfg * Dcfg;
    const long long N3D = (long long)Ncfg * D3cfg;
    const long long N4D = (long long)Ncfg * D4cfg;
    const long long HNN = (long long)NHcfg * NN;
    const float iscl = 0.125f;  // 1/sqrt(HD=64)
    const int RBN = Bcfg * Ncfg;

    // ================= FC branch =================
    // h = fc @ proj_w^T + proj_b
    gemm(stream, fc, NN, 0, Ncfg, proj_w, 0, 0, 1, Ncfg, proj_b,
         o_h, ND, 0, Dcfg, Ncfg, Dcfg, Ncfg, Bcfg, 1, 1.0f, 0);

    for (int l = 0; l < NLcfg; ++l) {
        const float* wqkv = eqkv_w + (size_t)l * D3cfg * Dcfg;
        const float* bqkv = eqkv_b + (size_t)l * D3cfg;
        // qkv = h @ wqkv^T + bqkv
        gemm(stream, o_h, ND, 0, Dcfg, wqkv, 0, 0, 1, Dcfg, bqkv,
             o_qkv, N3D, 0, D3cfg, Ncfg, D3cfg, Dcfg, Bcfg, 1, 1.0f, 0);
        // scores = Q @ K^T * (1/8)
        gemm(stream, o_qkv, N3D, HDcfg, D3cfg,
             o_qkv + Dcfg, N3D, HDcfg, 1, D3cfg, nullptr,
             o_big, HNN, NN, Ncfg, Ncfg, Ncfg, HDcfg, Bcfg, NHcfg, iscl, 0);
        softmax_rows<<<Bcfg * NHcfg * Ncfg, 32, 0, stream>>>(o_big, Bcfg * NHcfg * Ncfg, Ncfg);
        // ctx = attn @ V
        gemm(stream, o_big, HNN, NN, Ncfg,
             o_qkv + 2 * Dcfg, N3D, HDcfg, D3cfg, 1, nullptr,
             o_ctx, ND, HDcfg, Dcfg, Ncfg, HDcfg, Ncfg, Bcfg, NHcfg, 1.0f, 0);
        // out proj
        gemm(stream, o_ctx, ND, 0, Dcfg, eout_w + (size_t)l * Dcfg * Dcfg, 0, 0, 1, Dcfg,
             eout_b + (size_t)l * Dcfg, o_t2, ND, 0, Dcfg, Ncfg, Dcfg, Dcfg, Bcfg, 1, 1.0f, 0);
        // h = LN(h + attn)
        ln_add_rows<<<RBN, 32, 0, stream>>>(o_h, o_t2, en1_g + (size_t)l * Dcfg,
                                            en1_b + (size_t)l * Dcfg, o_h, RBN);
        // ffn
        gemm(stream, o_h, ND, 0, Dcfg, el1_w + (size_t)l * D4cfg * Dcfg, 0, 0, 1, Dcfg,
             el1_b + (size_t)l * D4cfg, o_t1, N4D, 0, D4cfg, Ncfg, D4cfg, Dcfg, Bcfg, 1, 1.0f, 1);
        gemm(stream, o_t1, N4D, 0, D4cfg, el2_w + (size_t)l * Dcfg * D4cfg, 0, 0, 1, D4cfg,
             el2_b + (size_t)l * Dcfg, o_t2, ND, 0, Dcfg, Ncfg, Dcfg, D4cfg, Bcfg, 1, 1.0f, 0);
        ln_add_rows<<<RBN, 32, 0, stream>>>(o_h, o_t2, en2_g + (size_t)l * Dcfg,
                                            en2_b + (size_t)l * Dcfg, o_h, RBN);
    }
    // Z_F = o_h

    // ================= SC (GCN) branch =================
    deg_dinv<<<(RBN + 255) / 256, 256, 0, stream>>>(sc, o_dnv, RBN);
    {
        long long tot = (long long)Bcfg * NN;
        build_anT<<<(unsigned)((tot + 255) / 256), 256, 0, stream>>>(sc, o_dnv, o_big, tot);
    }
    // X1 = sc @ gcn_w1^T
    gemm(stream, sc, NN, 0, Ncfg, gcn_w1, 0, 0, 1, Ncfg, nullptr,
         o_X, ND, 0, Dcfg, Ncfg, Dcfg, Ncfg, Bcfg, 1, 1.0f, 0);
    // h1 = relu(AnT @ X1 + b1)
    gemm(stream, o_big, NN, 0, Ncfg, o_X, ND, 0, Dcfg, 1, gcn_b1,
         o_h1, ND, 0, Dcfg, Ncfg, Dcfg, Ncfg, Bcfg, 1, 1.0f, 1);
    // X2 = h1 @ gcn_w2^T
    gemm(stream, o_h1, ND, 0, Dcfg, gcn_w2, 0, 0, 1, Dcfg, nullptr,
         o_X, ND, 0, Dcfg, Ncfg, Dcfg, Dcfg, Bcfg, 1, 1.0f, 0);
    // Z_S = AnT @ X2 + b2
    gemm(stream, o_big, NN, 0, Ncfg, o_X, ND, 0, Dcfg, 1, gcn_b2,
         o_ZS, ND, 0, Dcfg, Ncfg, Dcfg, Ncfg, Bcfg, 1, 1.0f, 0);

    // ================= cross attentions =================
    struct CA { const float* q; const float* kv; const float* w; const float* b;
                const float* ow; const float* ob; const float* ng; const float* nb; float* out; };
    CA cas[2] = {
        { o_h,  o_ZS, ca1_qkv_w, ca1_qkv_b, ca1_out_w, ca1_out_b, ca1_n_g, ca1_n_b, o_ZFc },
        { o_ZS, o_h,  ca2_qkv_w, ca2_qkv_b, ca2_out_w, ca2_out_b, ca2_n_g, ca2_n_b, o_ZSc },
    };
    for (int c = 0; c < 2; ++c) {
        const CA& a = cas[c];
        // Q/K/V projections into the [B,N,3D] qkv buffer (ldc = 3D)
        gemm(stream, a.q,  ND, 0, Dcfg, a.w,                       0, 0, 1, Dcfg, a.b,
             o_qkv,            N3D, 0, D3cfg, Ncfg, Dcfg, Dcfg, Bcfg, 1, 1.0f, 0);
        gemm(stream, a.kv, ND, 0, Dcfg, a.w + (size_t)Dcfg * Dcfg, 0, 0, 1, Dcfg, a.b + Dcfg,
             o_qkv + Dcfg,     N3D, 0, D3cfg, Ncfg, Dcfg, Dcfg, Bcfg, 1, 1.0f, 0);
        gemm(stream, a.kv, ND, 0, Dcfg, a.w + (size_t)2 * Dcfg * Dcfg, 0, 0, 1, Dcfg, a.b + 2 * Dcfg,
             o_qkv + 2 * Dcfg, N3D, 0, D3cfg, Ncfg, Dcfg, Dcfg, Bcfg, 1, 1.0f, 0);
        gemm(stream, o_qkv, N3D, HDcfg, D3cfg,
             o_qkv + Dcfg, N3D, HDcfg, 1, D3cfg, nullptr,
             o_big, HNN, NN, Ncfg, Ncfg, Ncfg, HDcfg, Bcfg, NHcfg, iscl, 0);
        softmax_rows<<<Bcfg * NHcfg * Ncfg, 32, 0, stream>>>(o_big, Bcfg * NHcfg * Ncfg, Ncfg);
        gemm(stream, o_big, HNN, NN, Ncfg,
             o_qkv + 2 * Dcfg, N3D, HDcfg, D3cfg, 1, nullptr,
             o_ctx, ND, HDcfg, Dcfg, Ncfg, HDcfg, Ncfg, Bcfg, NHcfg, 1.0f, 0);
        gemm(stream, o_ctx, ND, 0, Dcfg, a.ow, 0, 0, 1, Dcfg, a.ob,
             o_t2, ND, 0, Dcfg, Ncfg, Dcfg, Dcfg, Bcfg, 1, 1.0f, 0);
        ln_add_rows<<<RBN, 32, 0, stream>>>(a.q, o_t2, a.ng, a.nb, a.out, RBN);
    }

    // ================= classifier =================
    {
        long long tot = (long long)BND;
        avg2<<<(unsigned)((tot + 255) / 256), 256, 0, stream>>>(o_ZFc, o_ZSc, o_t1, tot);
    }
    const int ND_i = Ncfg * Dcfg;  // 102400
    // mid = relu(flat @ cls_w1^T + b1)   [B=64, 128]
    gemm(stream, o_t1, 0, 0, ND_i, cls_w1, 0, 0, 1, ND_i, cls_b1,
         o_cls, 0, 0, 128, Bcfg, 128, ND_i, 1, 1, 1.0f, 1);
    // logits = mid @ cls_w2^T + b2       [B=64, 2]
    gemm(stream, o_cls, 0, 0, 128, cls_w2, 0, 0, 1, 128, cls_b2,
         logits, 0, 0, NCcfg, Bcfg, NCcfg, 128, 1, 1, 1.0f, 0);
}
